// TransformerDecoder_21698174779948
// MI455X (gfx1250) — matrix operand order repaired and checked by reading, not compiled, not run on hardware
//
#include <hip/hip_runtime.h>

typedef __attribute__((ext_vector_type(16))) _Float16 v16h;
typedef __attribute__((ext_vector_type(8)))  _Float16 v8h;
typedef __attribute__((ext_vector_type(8)))  float    v8f;

// ---- problem constants ----
constexpr int kB    = 16;
constexpr int kLQ   = 300;
constexpr int kD    = 256;
constexpr int kNH   = 8;
constexpr int kHD   = 32;
constexpr int kFF   = 1024;
constexpr int kNBIN = 33;
constexpr int kSUMP = 12;
constexpr int kNC   = 80;
constexpr int kROWS = kB * kLQ;   // 4800 (multiple of 16)
constexpr int kSTOT = 8400;       // 80*80+40*40+20*20

enum { ACT_NONE = 0, ACT_RELU = 1, ACT_SILU = 2, ACT_SIGMOID = 3, ACT_CLIP10 = 4 };
enum { AM_PLAIN = 0, AM_ADD = 1, AM_CONCAT = 2 };

__device__ __forceinline__ v16h pack4(const float4 a, const float4 b,
                                      const float4 c, const float4 d) {
  v16h r;
  r[0]=(_Float16)a.x;  r[1]=(_Float16)a.y;  r[2]=(_Float16)a.z;  r[3]=(_Float16)a.w;
  r[4]=(_Float16)b.x;  r[5]=(_Float16)b.y;  r[6]=(_Float16)b.z;  r[7]=(_Float16)b.w;
  r[8]=(_Float16)c.x;  r[9]=(_Float16)c.y;  r[10]=(_Float16)c.z; r[11]=(_Float16)c.w;
  r[12]=(_Float16)d.x; r[13]=(_Float16)d.y; r[14]=(_Float16)d.z; r[15]=(_Float16)d.w;
  return r;
}
__device__ __forceinline__ v16h cat8(const v8h lo, const v8h hi) {
  v16h r;
  #pragma unroll
  for (int j = 0; j < 8; ++j) { r[j] = lo[j]; r[j + 8] = hi[j]; }
  return r;
}

// ============================================================================
// Generic WMMA GEMM: C[M,N] = act(Astage[M,K] @ W[K,N] + bias + R)
//   Astage: amode 0: A ; amode 1: A + A2 (same [M,K]) ; amode 2: [A | A2]
//           (two [M,256] halves, K must be 512)
// block = 128 threads (4 waves). Block tile 16x64, K-step 32, f16 in / f32 acc.
// qkv_store: remap [row=b*LQ+q, col=h*32+d] -> [((b*NH+h)*LQ+q)*32+d]
// ============================================================================
__global__ __launch_bounds__(128)
void gemm_wmma_kernel(const float* __restrict__ A, const float* __restrict__ A2,
                      const float* __restrict__ Wm,
                      const float* __restrict__ bias, const float* __restrict__ Rres,
                      float* __restrict__ C, int M, int N, int K,
                      int act, int qkv_store, int amode)
{
  __shared__ _Float16 As[16][48];    // rows padded to 96B: 16B-aligned chunks
  __shared__ _Float16 BsT[64][48];   // B transposed: BsT[n][k]
  const int tid  = threadIdx.x;
  const int wave = tid >> 5, lane = tid & 31;
  const int col  = lane & 15, hi = lane >> 4;
  const int m0   = blockIdx.y << 4;
  const int nblk = blockIdx.x << 6;
  const int n0   = nblk + (wave << 4);

  const int ar  = tid >> 3;          // A: row 0..15
  const int ac4 = (tid & 7) << 2;    // A: col chunk 0,4,..,28

  v8f acc = {0.f, 0.f, 0.f, 0.f, 0.f, 0.f, 0.f, 0.f};
  const int ksteps = (K + 31) >> 5;
  const bool fullN = (nblk + 64 <= N);

  for (int ks = 0; ks < ksteps; ++ks) {
    const int k0 = ks << 5;
    const bool fullK = (k0 + 32 <= K);
    // ---- stage A tile [16][32] -> f16 LDS ----
    if (fullK) {
      float4 av;
      if (amode == AM_CONCAT) {
        const int c = k0 + ac4;   // chunk never straddles the 256 boundary
        const float* src = (c < 256) ? (A + (size_t)(m0 + ar) * 256 + c)
                                     : (A2 + (size_t)(m0 + ar) * 256 + (c - 256));
        av = *(const float4*)src;
      } else {
        av = *(const float4*)(A + (size_t)(m0 + ar) * K + k0 + ac4);
        if (amode == AM_ADD) {
          const float4 a2 = *(const float4*)(A2 + (size_t)(m0 + ar) * K + k0 + ac4);
          av.x += a2.x; av.y += a2.y; av.z += a2.z; av.w += a2.w;
        }
      }
      As[ar][ac4 + 0] = (_Float16)av.x;
      As[ar][ac4 + 1] = (_Float16)av.y;
      As[ar][ac4 + 2] = (_Float16)av.z;
      As[ar][ac4 + 3] = (_Float16)av.w;
    } else {               // ragged K tail (only K=4 path)
      #pragma unroll
      for (int u = 0; u < 4; ++u) {
        const int c = ac4 + u;
        float v = 0.f;
        if (k0 + c < K) {
          v = A[(size_t)(m0 + ar) * K + k0 + c];
          if (amode == AM_ADD) v += A2[(size_t)(m0 + ar) * K + k0 + c];
        }
        As[ar][c] = (_Float16)v;
      }
    }
    // ---- stage B tile [32][64] -> f16 LDS transposed ----
    if (fullK && fullN) {
      #pragma unroll
      for (int i = 0; i < 4; ++i) {
        const int f  = tid + (i << 7);
        const int kr = f >> 4, c4 = (f & 15) << 2;
        const float4 bv = *(const float4*)(Wm + (size_t)(k0 + kr) * N + nblk + c4);
        BsT[c4 + 0][kr] = (_Float16)bv.x;
        BsT[c4 + 1][kr] = (_Float16)bv.y;
        BsT[c4 + 2][kr] = (_Float16)bv.z;
        BsT[c4 + 3][kr] = (_Float16)bv.w;
      }
    } else {
      #pragma unroll
      for (int i = 0; i < 4; ++i) {
        const int f  = tid + (i << 7);
        const int kr = f >> 4, c4 = (f & 15) << 2;
        #pragma unroll
        for (int u = 0; u < 4; ++u) {
          const int nn = nblk + c4 + u;
          float v = (k0 + kr < K && nn < N) ? Wm[(size_t)(k0 + kr) * N + nn] : 0.f;
          BsT[c4 + u][kr] = (_Float16)v;
        }
      }
    }
    // prefetch next weight K-tile while this one computes
    if (k0 + 32 < K)
      __builtin_prefetch(Wm + (size_t)(k0 + 32 + (tid >> 4)) * N + nblk + ((tid & 15) << 2), 0, 1);
    __syncthreads();
    if (n0 < N) {                                 // wave-uniform
      const v8h a_lo = *(const v8h*)&As[col][8 * hi];
      const v8h a_hi = *(const v8h*)&As[col][16 + 8 * hi];
      const v8h b_lo = *(const v8h*)&BsT[(wave << 4) + col][8 * hi];
      const v8h b_hi = *(const v8h*)&BsT[(wave << 4) + col][16 + 8 * hi];
      acc = __builtin_amdgcn_wmma_f32_16x16x32_f16(false, cat8(a_lo, a_hi),
                                                   false, cat8(b_lo, b_hi),
                                                   (short)0, acc, false, false);
    }
    __syncthreads();
  }
  if (n0 >= N) return;
  const int nn = n0 + col;
  const bool okn = nn < N;
  const float bv = (bias && okn) ? bias[nn] : 0.f;
  #pragma unroll
  for (int r = 0; r < 8; ++r) {
    const int mr = m0 + r + 8 * hi;
    float v = acc[r] + bv;
    if (Rres && okn) v += Rres[(size_t)mr * N + nn];
    if      (act == ACT_RELU)    v = fmaxf(v, 0.f);
    else if (act == ACT_SILU)    v = v / (1.f + __expf(-v));
    else if (act == ACT_SIGMOID) v = 1.f / (1.f + __expf(-v));
    else if (act == ACT_CLIP10)  v = fminf(fmaxf(v, -10.f), 10.f);
    if (okn) {
      if (!qkv_store) {
        C[(size_t)mr * N + nn] = v;
      } else {
        const int b = mr / kLQ, q = mr % kLQ;
        const int h = nn >> 5, d = nn & 31;
        C[(((size_t)b * kNH + h) * kLQ + q) * kHD + d] = v;
      }
    }
  }
}

// ============================================================================
// Fused self-attention. One wave per (b*NH, 16-query tile).
// Q,K,V: [B*NH, LQ, 32].  O: [B, LQ, 256] with col = h*32+d.
// ============================================================================
__global__ __launch_bounds__(32)
void attn_wmma_kernel(const float* __restrict__ Qp, const float* __restrict__ Kp,
                      const float* __restrict__ Vp, float* __restrict__ O)
{
  __shared__ float S[16][320];
  const int lane = threadIdx.x & 31;
  const int col = lane & 15, hi = lane >> 4;
  const int qt = blockIdx.x;        // 0..18
  const int bh = blockIdx.y;        // 0..127
  const int b = bh >> 3, h = bh & 7;
  const int m0 = qt << 4;

  v16h af;
  {
    int qr = m0 + col; if (qr > kLQ - 1) qr = kLQ - 1;
    const float* qrow = Qp + ((size_t)bh * kLQ + qr) * kHD + 8 * hi;
    af = pack4(*(const float4*)(qrow), *(const float4*)(qrow + 4),
               *(const float4*)(qrow + 16), *(const float4*)(qrow + 20));
  }
  const float scale = 0.17677669529663687f; // 32^-0.5
  for (int kt = 0; kt < 19; ++kt) {
    int key = (kt << 4) + col; if (key > kLQ - 1) key = kLQ - 1;
    const float* krow = Kp + ((size_t)bh * kLQ + key) * kHD + 8 * hi;
    const v16h bf = pack4(*(const float4*)(krow), *(const float4*)(krow + 4),
                          *(const float4*)(krow + 16), *(const float4*)(krow + 20));
    v8f c = {0.f, 0.f, 0.f, 0.f, 0.f, 0.f, 0.f, 0.f};
    c = __builtin_amdgcn_wmma_f32_16x16x32_f16(false, af, false, bf,
                                               (short)0, c, false, false);
    #pragma unroll
    for (int r = 0; r < 8; ++r) S[r + 8 * hi][(kt << 4) + col] = c[r] * scale;
  }
  for (int t = lane; t < 16 * 20; t += 32) {   // keys 300..319 -> -inf
    int r = t / 20, cc = 300 + (t % 20);
    S[r][cc] = -1e30f;
  }
  __syncthreads();
  { // softmax over 300 keys; lane (row=col, part=hi) scans 160 entries
    const int row = col, part = hi;
    float mx = -1e30f;
    for (int i = 0; i < 160; ++i) mx = fmaxf(mx, S[row][part * 160 + i]);
    mx = fmaxf(mx, __shfl_xor(mx, 16));
    float sum = 0.f;
    for (int i = 0; i < 160; ++i) {
      float e = __expf(S[row][part * 160 + i] - mx);
      S[row][part * 160 + i] = e; sum += e;
    }
    sum += __shfl_xor(sum, 16);
    const float inv = 1.f / sum;
    for (int i = 0; i < 160; ++i) S[row][part * 160 + i] *= inv;
  }
  __syncthreads();
  v8f c0 = {0.f, 0.f, 0.f, 0.f, 0.f, 0.f, 0.f, 0.f};
  v8f c1 = {0.f, 0.f, 0.f, 0.f, 0.f, 0.f, 0.f, 0.f};
  for (int ck = 0; ck < 10; ++ck) {    // P[16x320] @ V[320x32]
    const int k0 = ck << 5;
    const float* srow = &S[col][k0 + 8 * hi];
    const v16h pa = pack4(*(const float4*)(srow), *(const float4*)(srow + 4),
                          *(const float4*)(srow + 16), *(const float4*)(srow + 20));
    v16h bv0, bv1;
    #pragma unroll
    for (int j = 0; j < 16; ++j) {
      int key = k0 + 8 * hi + (j < 8 ? j : j + 8); if (key > kLQ - 1) key = kLQ - 1; // P==0 there
      const float* vrow = Vp + ((size_t)bh * kLQ + key) * kHD;
      bv0[j] = (_Float16)vrow[col];
      bv1[j] = (_Float16)vrow[16 + col];
    }
    c0 = __builtin_amdgcn_wmma_f32_16x16x32_f16(false, pa, false, bv0,
                                                (short)0, c0, false, false);
    c1 = __builtin_amdgcn_wmma_f32_16x16x32_f16(false, pa, false, bv1,
                                                (short)0, c1, false, false);
  }
  #pragma unroll
  for (int r = 0; r < 8; ++r) {
    const int q = m0 + r + 8 * hi;
    if (q < kLQ) {
      float* orow = O + ((size_t)b * kLQ + q) * kD + h * kHD;
      orow[col] = c0[r];
      orow[16 + col] = c1[r];
    }
  }
}

// ============================================================================
// Deformable attention sampling. One wave per (b,q,h); lane = channel d.
// ============================================================================
__global__ __launch_bounds__(256)
void deform_kernel(const float* __restrict__ mem, const float* __restrict__ refp,
                   const float* __restrict__ offs, const float* __restrict__ aw,
                   float* __restrict__ t2)
{
  int gw = (int)((blockIdx.x * (size_t)blockDim.x + threadIdx.x) >> 5);
  int lane = threadIdx.x & 31;
  if (gw >= kROWS * kNH) return;
  int h = gw & 7; int bq = gw >> 3;
  int b = bq / kLQ;
  const float* rp = refp + (size_t)bq * 4;
  float cx = rp[0], cy = rp[1], cw = rp[2], ch = rp[3];
  const float* of = offs + ((size_t)bq * kNH + h) * (kSUMP * 2);
  const float* ap = aw + ((size_t)bq * kNH + h) * kSUMP;
  const float* mbase = mem + (size_t)b * kSTOT * kD + h * kHD + lane;
  float acc = 0.f;
  int s0 = 0;
  const int Wl3[3] = {80, 40, 20};
  #pragma unroll
  for (int lvl = 0; lvl < 3; ++lvl) {
    int Wl = Wl3[lvl];
    float fW = (float)Wl;
    #pragma unroll
    for (int pp = 0; pp < 4; ++pp) {
      int p = lvl * 4 + pp;
      // offset = off * (1/NPTS) * wh * OFFSET_SCALE = off * 0.125 * wh
      float lx = cx + of[2 * p]     * 0.125f * cw;
      float ly = cy + of[2 * p + 1] * 0.125f * ch;
      float x = lx * fW - 0.5f;   // align_corners=False ; H == W per level
      float y = ly * fW - 0.5f;
      float x0f = floorf(x), y0f = floorf(y);
      float wx = x - x0f, wy = y - y0f;
      int x0 = (int)x0f, y0 = (int)y0f;
      float w = ap[p];
      float v = 0.f;
      #pragma unroll
      for (int dy = 0; dy < 2; ++dy) {
        int yi = y0 + dy;
        if (yi < 0 || yi >= Wl) continue;
        float wyv = dy ? wy : (1.f - wy);
        #pragma unroll
        for (int dx = 0; dx < 2; ++dx) {
          int xi = x0 + dx;
          if (xi < 0 || xi >= Wl) continue;
          float wxv = dx ? wx : (1.f - wx);
          v += wxv * wyv * mbase[(size_t)(s0 + yi * Wl + xi) * kD];
        }
      }
      acc += w * v;
    }
    s0 += Wl * Wl;
  }
  t2[(size_t)bq * kD + h * kHD + lane] = acc;
}

// ============================================================================
// LayerNorm family (one wave per 256-wide row)
// ============================================================================
__global__ __launch_bounds__(256)
void ln_kernel(const float* __restrict__ x, const float* __restrict__ res,
               const float* __restrict__ g, const float* __restrict__ b,
               float* __restrict__ y, int clip)
{
  int wid = (int)((blockIdx.x * (size_t)blockDim.x + threadIdx.x) >> 5);
  int lane = threadIdx.x & 31;
  if (wid >= kROWS) return;
  float v[8]; float s = 0.f;
  #pragma unroll
  for (int i = 0; i < 8; ++i) {
    int c = lane + (i << 5);
    float t = x[(size_t)wid * kD + c];
    if (res) t += res[(size_t)wid * kD + c];
    if (clip) t = fminf(fmaxf(t, -65504.f), 65504.f);
    v[i] = t; s += t;
  }
  #pragma unroll
  for (int o = 16; o; o >>= 1) s += __shfl_xor(s, o);
  float mu = s * (1.f / 256.f);
  float var = 0.f;
  #pragma unroll
  for (int i = 0; i < 8; ++i) { float d = v[i] - mu; var += d * d; }
  #pragma unroll
  for (int o = 16; o; o >>= 1) var += __shfl_xor(var, o);
  float inv = rsqrtf(var * (1.f / 256.f) + 1e-5f);
  #pragma unroll
  for (int i = 0; i < 8; ++i) {
    int c = lane + (i << 5);
    y[(size_t)wid * kD + c] = (v[i] - mu) * inv * g[c] + b[c];
  }
}

__global__ __launch_bounds__(256)
void ln_gate_kernel(const float* __restrict__ g12, const float* __restrict__ xo,
                    const float* __restrict__ t2, const float* __restrict__ g,
                    const float* __restrict__ b, float* __restrict__ y)
{
  int wid = (int)((blockIdx.x * (size_t)blockDim.x + threadIdx.x) >> 5);
  int lane = threadIdx.x & 31;
  if (wid >= kROWS) return;
  float v[8]; float s = 0.f;
  #pragma unroll
  for (int i = 0; i < 8; ++i) {
    int c = lane + (i << 5);
    float t = g12[(size_t)wid * 512 + c]       * xo[(size_t)wid * kD + c]
            + g12[(size_t)wid * 512 + 256 + c] * t2[(size_t)wid * kD + c];
    v[i] = t; s += t;
  }
  #pragma unroll
  for (int o = 16; o; o >>= 1) s += __shfl_xor(s, o);
  float mu = s * (1.f / 256.f);
  float var = 0.f;
  #pragma unroll
  for (int i = 0; i < 8; ++i) { float d = v[i] - mu; var += d * d; }
  #pragma unroll
  for (int o = 16; o; o >>= 1) var += __shfl_xor(var, o);
  float inv = rsqrtf(var * (1.f / 256.f) + 1e-5f);
  #pragma unroll
  for (int i = 0; i < 8; ++i) {
    int c = lane + (i << 5);
    y[(size_t)wid * kD + c] = (v[i] - mu) * inv * g[c] + b[c];
  }
}

// ============================================================================
// small elementwise / decode kernels
// ============================================================================
__global__ void ew_copy(const float* __restrict__ a, float* __restrict__ y, int n) {
  int i = blockIdx.x * blockDim.x + threadIdx.x;
  if (i < n) y[i] = a[i];
}
__global__ void ew_zero(float* __restrict__ y, int n) {
  int i = blockIdx.x * blockDim.x + threadIdx.x;
  if (i < n) y[i] = 0.f;
}
__global__ void ew_sigmoid(const float* __restrict__ a, float* __restrict__ y, int n) {
  int i = blockIdx.x * blockDim.x + threadIdx.x;
  if (i < n) y[i] = 1.f / (1.f + __expf(-a[i]));
}
__global__ void softmax12_kernel(float* __restrict__ aw, int n) {
  int i = blockIdx.x * blockDim.x + threadIdx.x;
  if (i >= n) return;
  float* p = aw + (size_t)i * 12;
  float mx = p[0];
  #pragma unroll
  for (int j = 1; j < 12; ++j) mx = fmaxf(mx, p[j]);
  float e[12]; float s = 0.f;
  #pragma unroll
  for (int j = 0; j < 12; ++j) { e[j] = __expf(p[j] - mx); s += e[j]; }
  float inv = 1.f / s;
  #pragma unroll
  for (int j = 0; j < 12; ++j) p[j] = e[j] * inv;
}
__global__ void prebb_kernel(const float* __restrict__ raw, const float* __restrict__ refp,
                             float* __restrict__ refi, int n) {
  int i = blockIdx.x * blockDim.x + threadIdx.x;
  if (i >= n) return;
  float x = fminf(fmaxf(refp[i], 1e-5f), 1.f - 1e-5f);
  float is = logf(x) - log1pf(-x);
  float v = raw[i] + is;
  refi[i] = 1.f / (1.f + __expf(-v));
}
__global__ void proj_init_kernel(float* __restrict__ proj) {
  int i = threadIdx.x;
  if (i >= kNBIN) return;
  const float step = powf(3.0f, 1.0f / 15.0f);
  float v;
  if (i == 0)       v = -4.0f;
  else if (i < 16)  v = -powf(step, (float)(16 - i)) + 1.0f;
  else if (i == 16) v = 0.f;
  else if (i < 32)  v =  powf(step, (float)(i - 16)) - 1.0f;
  else              v = 4.0f;
  proj[i] = v;
}
// corners -> softmax @ proj -> distance2bbox(ref_init) -> new ref_pts
__global__ void fdr_kernel(const float* __restrict__ corners, const float* __restrict__ proj,
                           const float* __restrict__ refi, float* __restrict__ refp_out,
                           float* __restrict__ bb_out)
{
  int idx = blockIdx.x * blockDim.x + threadIdx.x;
  if (idx >= kROWS) return;
  const float* cr = corners + (size_t)idx * (4 * kNBIN);
  float dist[4];
  #pragma unroll
  for (int sd = 0; sd < 4; ++sd) {
    const float* c4 = cr + sd * kNBIN;
    float mx = -1e30f;
    for (int j = 0; j < kNBIN; ++j) mx = fmaxf(mx, c4[j]);
    float s = 0.f, dot = 0.f;
    for (int j = 0; j < kNBIN; ++j) {
      float e = __expf(c4[j] - mx);
      s += e; dot += e * proj[j];
    }
    dist[sd] = dot / s;
  }
  const float* pt = refi + (size_t)idx * 4;
  const float inv_rs = 0.25f, half = 2.0f;
  float x1 = pt[0] - (half + dist[0]) * (pt[2] * inv_rs);
  float y1 = pt[1] - (half + dist[1]) * (pt[3] * inv_rs);
  float x2 = pt[0] + (half + dist[2]) * (pt[2] * inv_rs);
  float y2 = pt[1] + (half + dist[3]) * (pt[3] * inv_rs);
  float o0 = (x1 + x2) * 0.5f, o1 = (y1 + y2) * 0.5f, o2 = x2 - x1, o3 = y2 - y1;
  refp_out[(size_t)idx * 4 + 0] = o0;
  refp_out[(size_t)idx * 4 + 1] = o1;
  refp_out[(size_t)idx * 4 + 2] = o2;
  refp_out[(size_t)idx * 4 + 3] = o3;
  if (bb_out) {
    bb_out[(size_t)idx * 84 + 0] = o0;
    bb_out[(size_t)idx * 84 + 1] = o1;
    bb_out[(size_t)idx * 84 + 2] = o2;
    bb_out[(size_t)idx * 84 + 3] = o3;
  }
}
// LQE head: one wave per row.
__global__ __launch_bounds__(256)
void lqe_kernel(const float* __restrict__ scores, const float* __restrict__ corners,
                const float* __restrict__ w1, const float* __restrict__ b1,
                const float* __restrict__ w2, const float* __restrict__ b2,
                float* __restrict__ outp)
{
  __shared__ float stat[8][20];
  int lwave = threadIdx.x >> 5, lane = threadIdx.x & 31;
  int row = blockIdx.x * 8 + lwave;   // grid = 600 exact
  const float* cr = corners + (size_t)row * (4 * kNBIN);
  if (lane < 4) {
    const float* c4 = cr + lane * kNBIN;
    float mx = -1e30f;
    for (int j = 0; j < kNBIN; ++j) mx = fmaxf(mx, c4[j]);
    float sum = 0.f;
    for (int j = 0; j < kNBIN; ++j) sum += __expf(c4[j] - mx);
    float invs = 1.f / sum;
    float t0 = -1.f, t1 = -1.f, t2 = -1.f, t3 = -1.f;
    for (int j = 0; j < kNBIN; ++j) {
      float p = __expf(c4[j] - mx) * invs;
      if (p > t0)      { t3 = t2; t2 = t1; t1 = t0; t0 = p; }
      else if (p > t1) { t3 = t2; t2 = t1; t1 = p; }
      else if (p > t2) { t3 = t2; t2 = p; }
      else if (p > t3) { t3 = p; }
    }
    stat[lwave][lane * 5 + 0] = t0;
    stat[lwave][lane * 5 + 1] = t1;
    stat[lwave][lane * 5 + 2] = t2;
    stat[lwave][lane * 5 + 3] = t3;
    stat[lwave][lane * 5 + 4] = (t0 + t1 + t2 + t3) * 0.25f;
  }
  __syncthreads();
  float acc = 0.f;
  #pragma unroll
  for (int t = 0; t < 2; ++t) {
    int hx = lane + (t << 5);
    float hsum = b1[hx];
    #pragma unroll
    for (int k = 0; k < 20; ++k) hsum += stat[lwave][k] * w1[k * 64 + hx];
    float hs = hsum / (1.f + __expf(-hsum));   // silu
    acc += hs * w2[hx];
  }
  #pragma unroll
  for (int o = 16; o; o >>= 1) acc += __shfl_xor(acc, o);
  float lq = acc + b2[0];
  for (int c = lane; c < kNC; c += 32)
    outp[(size_t)row * 84 + 4 + c] = scores[(size_t)row * kNC + c] + lq;
}

// ============================================================================
// host orchestration
// ============================================================================
static inline void launch_gemm(hipStream_t s, const float* A, const float* A2, int amode,
                               const float* Wm, const float* bias, const float* Rres,
                               float* C, int M, int N, int K, int act, int qkv) {
  dim3 grid((N + 63) / 64, M / 16);
  gemm_wmma_kernel<<<grid, 128, 0, s>>>(A, A2, Wm, bias, Rres, C, M, N, K, act, qkv, amode);
}
static inline int gblk(int n) { return (n + 255) / 256; }

extern "C" void kernel_launch(void* const* d_in, const int* in_sizes, int n_in,
                              void* d_out, int out_size, void* d_ws, size_t ws_size,
                              hipStream_t stream)
{
  (void)in_sizes; (void)n_in; (void)out_size; (void)ws_size;
  auto F = [&](int i) { return (const float*)d_in[i]; };

  // ---- workspace layout (floats); every block is a multiple of 4 floats ----
  float* p = (float*)d_ws;
  auto take = [&](size_t n) { float* r = p; p += n; return r; };
  float* proj    = take(64);
  float* refp    = take((size_t)kROWS * 4);
  float* refi    = take((size_t)kROWS * 4);
  float* sm4     = take((size_t)kROWS * 4);
  float* outb    = take((size_t)kROWS * kD);
  float* outprev = take((size_t)kROWS * kD);
  float* cornA   = take((size_t)kROWS * 132);
  float* cornB   = take((size_t)kROWS * 132);
  float* qp      = take((size_t)kROWS * kD);
  float* hid     = take((size_t)kROWS * kFF);
  float* hid2    = take((size_t)kROWS * kD);
  float* qkb     = take((size_t)kROWS * kD);
  float* kkb     = take((size_t)kROWS * kD);
  float* vvb     = take((size_t)kROWS * kD);
  float* attno   = take((size_t)kROWS * kD);
  float* t2b     = take((size_t)kROWS * kD);
  float* offsb   = take((size_t)kROWS * 192);
  float* anwb    = take((size_t)kROWS * 96);
  float* scoresb = take((size_t)kROWS * kNC);
  float* g12     = qkb;   // reuse qkb+kkb (512/row contiguous; free after attn)

  // ---- init ----
  proj_init_kernel<<<1, 64, 0, stream>>>(proj);
  ew_sigmoid<<<gblk(kROWS * 4), 256, 0, stream>>>(F(1), refp, kROWS * 4);
  ew_copy<<<gblk(kROWS * kD), 256, 0, stream>>>(F(0), outb, kROWS * kD);
  ew_zero<<<gblk(kROWS * kD), 256, 0, stream>>>(outprev, kROWS * kD);
  ew_zero<<<gblk(kROWS * 132), 256, 0, stream>>>(cornB, kROWS * 132);

  const float* mem = F(2);
  for (int i = 0; i < 6; ++i) {
    float* cornCur  = (i & 1) ? cornB : cornA;
    float* cornPrev = (i & 1) ? cornA : cornB;
    const float* qw  = F(3)  + (size_t)i * kD * kD;  const float* qb_  = F(4)  + (size_t)i * kD;
    const float* kw  = F(5)  + (size_t)i * kD * kD;  const float* kb_  = F(6)  + (size_t)i * kD;
    const float* vw  = F(7)  + (size_t)i * kD * kD;  const float* vb_  = F(8)  + (size_t)i * kD;
    const float* ow  = F(9)  + (size_t)i * kD * kD;  const float* ob_  = F(10) + (size_t)i * kD;
    const float* n1g = F(11) + (size_t)i * kD;       const float* n1b  = F(12) + (size_t)i * kD;
    const float* ofw = F(13) + (size_t)i * kD * 192; const float* ofb  = F(14) + (size_t)i * 192;
    const float* aww = F(15) + (size_t)i * kD * 96;  const float* awb  = F(16) + (size_t)i * 96;
    const float* gw  = F(17) + (size_t)i * 512 * 512;const float* gb_  = F(18) + (size_t)i * 512;
    const float* gng = F(19) + (size_t)i * kD;       const float* gnb  = F(20) + (size_t)i * kD;
    const float* f1w = F(21) + (size_t)i * kD * kFF; const float* f1b  = F(22) + (size_t)i * kFF;
    const float* f2w = F(23) + (size_t)i * kFF * kD; const float* f2b  = F(24) + (size_t)i * kD;
    const float* n3g = F(25) + (size_t)i * kD;       const float* n3b  = F(26) + (size_t)i * kD;
    const float* b1w = F(27) + (size_t)i * kD * kD;  const float* b1b  = F(28) + (size_t)i * kD;
    const float* b2w = F(29) + (size_t)i * kD * kD;  const float* b2b  = F(30) + (size_t)i * kD;
    const float* b3w = F(31) + (size_t)i * kD * 132; const float* b3b  = F(32) + (size_t)i * 132;

    // query pos embedding: qp = clip(mlp2(ref_pts), +/-10)
    launch_gemm(stream, refp, nullptr, AM_PLAIN, F(39), F(40), nullptr, hid, kROWS, 512, 4, ACT_SILU, 0);
    launch_gemm(stream, hid, nullptr, AM_PLAIN, F(41), F(42), nullptr, qp, kROWS, kD, 512, ACT_CLIP10, 0);

    // --- self attention (q/k input = output+qp fused into GEMM A-stage) ---
    launch_gemm(stream, outb, qp, AM_ADD, qw, qb_, nullptr, qkb, kROWS, kD, kD, ACT_NONE, 1);
    launch_gemm(stream, outb, qp, AM_ADD, kw, kb_, nullptr, kkb, kROWS, kD, kD, ACT_NONE, 1);
    launch_gemm(stream, outb, nullptr, AM_PLAIN, vw, vb_, nullptr, vvb, kROWS, kD, kD, ACT_NONE, 1);
    attn_wmma_kernel<<<dim3(19, kB * kNH), 32, 0, stream>>>(qkb, kkb, vvb, attno);
    launch_gemm(stream, attno, nullptr, AM_PLAIN, ow, ob_, nullptr, t2b, kROWS, kD, kD, ACT_NONE, 0);
    ln_kernel<<<600, 256, 0, stream>>>(outb, t2b, n1g, n1b, outb, 0);

    // --- deformable cross attention ---
    launch_gemm(stream, outb, qp, AM_ADD, ofw, ofb, nullptr, offsb, kROWS, 192, kD, ACT_NONE, 0);
    launch_gemm(stream, outb, qp, AM_ADD, aww, awb, nullptr, anwb, kROWS, 96, kD, ACT_NONE, 0);
    softmax12_kernel<<<gblk(kROWS * kNH), 256, 0, stream>>>(anwb, kROWS * kNH);
    deform_kernel<<<kROWS, 256, 0, stream>>>(mem, refp, offsb, anwb, t2b);

    // --- gateway (concat fused into GEMM A-stage) ---
    launch_gemm(stream, outb, t2b, AM_CONCAT, gw, gb_, nullptr, g12, kROWS, 512, 512, ACT_SIGMOID, 0);
    ln_gate_kernel<<<600, 256, 0, stream>>>(g12, outb, t2b, gng, gnb, outb);

    // --- FFN ---
    launch_gemm(stream, outb, nullptr, AM_PLAIN, f1w, f1b, nullptr, hid, kROWS, kFF, kD, ACT_RELU, 0);
    launch_gemm(stream, hid, nullptr, AM_PLAIN, f2w, f2b, nullptr, t2b, kROWS, kD, kFF, ACT_NONE, 0);
    ln_kernel<<<600, 256, 0, stream>>>(outb, t2b, n3g, n3b, outb, 1);

    // --- layer-0 pre-bbox head -> ref_init ---
    if (i == 0) {
      launch_gemm(stream, outb, nullptr, AM_PLAIN, F(43), F(44), nullptr, hid, kROWS, kD, kD, ACT_SILU, 0);
      launch_gemm(stream, hid, nullptr, AM_PLAIN, F(45), F(46), nullptr, hid2, kROWS, kD, kD, ACT_SILU, 0);
      launch_gemm(stream, hid2, nullptr, AM_PLAIN, F(47), F(48), nullptr, sm4, kROWS, 4, kD, ACT_NONE, 0);
      prebb_kernel<<<gblk(kROWS * 4), 256, 0, stream>>>(sm4, refp, refi, kROWS * 4);
    }

    // --- FDR box refinement (A = output + out_prev fused) ---
    launch_gemm(stream, outb, outprev, AM_ADD, b1w, b1b, nullptr, hid, kROWS, kD, kD, ACT_SILU, 0);
    launch_gemm(stream, hid, nullptr, AM_PLAIN, b2w, b2b, nullptr, hid2, kROWS, kD, kD, ACT_SILU, 0);
    launch_gemm(stream, hid2, nullptr, AM_PLAIN, b3w, b3b, cornPrev, cornCur, kROWS, 132, kD, ACT_NONE, 0);
    fdr_kernel<<<gblk(kROWS), 256, 0, stream>>>(cornCur, proj, refi, refp,
                                                (i == 5) ? (float*)d_out : nullptr);

    // --- eval head at last layer ---
    if (i == 5) {
      const float* scw = F(33) + (size_t)5 * kD * kNC;
      const float* scb = F(34) + (size_t)5 * kNC;
      launch_gemm(stream, outb, nullptr, AM_PLAIN, scw, scb, nullptr, scoresb, kROWS, kNC, kD, ACT_NONE, 0);
      lqe_kernel<<<600, 256, 0, stream>>>(scoresb, cornCur,
                                          F(35) + (size_t)5 * 20 * 64, F(36) + (size_t)5 * 64,
                                          F(37) + (size_t)5 * 64, F(38) + (size_t)5,
                                          (float*)d_out);
    }

    // out_prev = output (snapshot for next layer)
    ew_copy<<<gblk(kROWS * kD), 256, 0, stream>>>(outb, outprev, kROWS * kD);
  }
}
